// MS_SSA_dend_Conv_int_20117626814982
// MI455X (gfx1250) — matrix operand order repaired and checked by reading, not compiled, not run on hardware
//
#include <hip/hip_runtime.h>

// ---------------------------------------------------------------------------
// MS-SSA spiking self-attention block for MI455X (gfx1250, wave32, WMMA).
// T=4 B=16 C=512 N=H*W=256. All GEMMs via v_wmma_f32_16x16x32_f16.
// Spikes are {0,1} -> exact in f16; weights rounded to f16 (bandwidth win).
// GEMM blocks stage the shared activation B-panel in LDS (1 read instead of 8).
// ---------------------------------------------------------------------------

#define TDIM 4
#define BDIM 16
#define CDIM 512
#define NDIM 256
#define HEADS 8
#define DH 64
#define BCN  (BDIM * CDIM * NDIM)   // 2,097,152 (per timestep)
#define TBCN (TDIM * BCN)           // 8,388,608
#define BN_EPS 1e-5f

typedef __attribute__((ext_vector_type(16))) _Float16 v16h;
typedef __attribute__((ext_vector_type(8)))  _Float16 v8h;
typedef __attribute__((ext_vector_type(8)))  float    v8f;

// Workspace layout (bytes)
static const size_t XS_PACK_OFF = 0;            // 16,777,216  f16 spikes, B-tile order
static const size_t WPACK_OFF   = 16777216;     //  2,097,152  4x(512x512) f16, A-tile order
static const size_t QCUR_OFF    = 18874368;     // 33,554,432  f32 q branch current
static const size_t KCUR_OFF    = 52428800;     // 33,554,432  f32 k current (reused as OUT_PACK)
static const size_t VCUR_OFF    = 85983232;     // 33,554,432  f32 v current
static const size_t QS_OFF      = 119537664;    // 16,777,216  f16 q spikes (t,b,c,n)
static const size_t KS_OFF      = 136314880;    // 16,777,216  f16 k spikes
static const size_t VS_OFF      = 153092096;    // 16,777,216  f16 v spikes
static const size_t KVCUR_OFF   = 169869312;    //    131,072  f32 kv current (t*8192 + b*512+c)
static const size_t KVS_OFF     = 170000384;    //    131,072  f32 kv spikes
static const size_t OUTPACK_OFF = KCUR_OFF;     // alias: k-current dead after LIF

// B-matrix (32x16, f16) lane layout for wmma_f32_16x16x32_f16:
// lane = ((Kin/8)&1)*16 + col ; half j of v16h = element K = 8*(lane/16)+(j<8?j:j+8), N = lane%16
__device__ __forceinline__ int pack_b_idx(int tb, int c, int n) {
    int kk = c >> 5, cr = c & 31;
    int grp = (cr >> 3) & 1, j = (cr & 7) + ((cr >> 4) << 3);
    int nt = n >> 4, col = n & 15;
    int lane = grp * 16 + col;
    return (((tb * 16 + nt) * 16 + kk) * 32 + lane) * 16 + j;
}

// --- K0: convert + swizzle 4 weight matrices into WMMA A-tile order --------
// A (16x32 f16): lane<16 holds row=lane, halves 0..7 -> K 0..7, 8..15 -> K16..23
//                lane>=16 same row, K+8 on both ranges.
__global__ void k_pack_weights(const float* __restrict__ wq,
                               const float* __restrict__ wk,
                               const float* __restrict__ wv,
                               const float* __restrict__ wp,
                               _Float16* __restrict__ wpack) {
    int gid = blockIdx.x * blockDim.x + threadIdx.x;   // 4 * 2^18 threads
    int w   = gid >> 18;
    int pid = gid & 262143;
    int j    = pid & 15;
    int lane = (pid >> 4) & 31;
    int kk   = (pid >> 9) & 15;
    int mt   = pid >> 13;
    int row  = mt * 16 + (lane & 15);
    int kbas = (j < 8) ? j : (j + 8);
    int K    = kk * 32 + kbas + (lane >> 4) * 8;
    const float* src = (w == 0) ? wq : (w == 1) ? wk : (w == 2) ? wv : wp;
    wpack[gid] = (_Float16)src[row * CDIM + K];
}

// --- K1: shortcut LIF on x, emit spikes pre-swizzled into B-tile order -----
__global__ void k_lif_shortcut(const float* __restrict__ x,
                               _Float16* __restrict__ xs_pack) {
    int i = blockIdx.x * blockDim.x + threadIdx.x;     // over (b,c,n) = 2^21
    int b = i >> 17, c = (i >> 8) & 511, n = i & 255;
    float v = 0.f;
#pragma unroll
    for (int t = 0; t < TDIM; ++t) {
        float xt = x[t * BCN + i];
        v += (xt - v) * 0.5f;                          // charge, tau=2
        float s = (v >= 1.0f) ? 1.0f : 0.0f;           // fire
        v *= (1.0f - s);                               // hard reset
        xs_pack[pack_b_idx(t * BDIM + b, c, n)] = (_Float16)s;
    }
}

// --- K2: fused q/k/v GEMM, B-panel staged in LDS, BN folded in epilogue ----
// Block = 8 waves; block owns one (tb,nt,mt-group); wave w -> mt = mgrp*8+w.
// All 8 waves consume the same 16KB xs B-panel -> load it into LDS once.
__global__ void __launch_bounds__(256)
k_gemm_qkv(const _Float16* __restrict__ xsp,
           const _Float16* __restrict__ wpack,
           const float* __restrict__ qg, const float* __restrict__ qb,
           const float* __restrict__ qm, const float* __restrict__ qv,
           const float* __restrict__ kg, const float* __restrict__ kb,
           const float* __restrict__ km, const float* __restrict__ kv_,
           const float* __restrict__ vg, const float* __restrict__ vb,
           const float* __restrict__ vm, const float* __restrict__ vv,
           float* __restrict__ qcur, float* __restrict__ kcur,
           float* __restrict__ vcur) {
    __shared__ _Float16 Bs[16 * 512];                  // 16 K-chunks x 512 halfs = 16KB
    int lane = threadIdx.x & 31;
    int wave = threadIdx.x >> 5;
    int blk  = blockIdx.x;                             // 64*16*4 = 4096 blocks
    int mgrp = blk & 3, nt = (blk >> 2) & 15, tb = blk >> 6;
    int mt   = mgrp * 8 + wave;

    // cooperative stage: 512 contiguous v16h (16KB) global -> LDS
    {
        const v16h* src = (const v16h*)xsp + (size_t)((tb * 16 + nt) * 16) * 32;
        v16h* dst = (v16h*)Bs;
        dst[threadIdx.x]       = src[threadIdx.x];
        dst[threadIdx.x + 256] = src[threadIdx.x + 256];
    }
    __syncthreads();

    const v16h* Bl = (const v16h*)Bs + lane;
    const v16h* Aq = (const v16h*)wpack + (size_t)(mt * 16) * 32 + lane;
    const v16h* Ak = Aq + 16384;                       // 262144 halfs per weight
    const v16h* Av = Ak + 16384;

    v8f aq = {}, ak = {}, av = {};
#pragma unroll 4
    for (int kk = 0; kk < 16; ++kk) {
        v16h btile = Bl[kk * 32];                      // ds_load from LDS panel
        if (kk + 1 < 16) __builtin_prefetch((const void*)(Aq + (kk + 1) * 32), 0, 1);
        v16h a0 = Aq[kk * 32], a1 = Ak[kk * 32], a2 = Av[kk * 32];
        aq = __builtin_amdgcn_wmma_f32_16x16x32_f16(false, a0, false, btile, (short)0, aq, false, false);
        ak = __builtin_amdgcn_wmma_f32_16x16x32_f16(false, a1, false, btile, (short)0, ak, false, false);
        av = __builtin_amdgcn_wmma_f32_16x16x32_f16(false, a2, false, btile, (short)0, av, false, false);
    }
    int col = lane & 15, hi = lane >> 4;
    int n = nt * 16 + col;
#pragma unroll
    for (int r = 0; r < 8; ++r) {
        int o = mt * 16 + r + hi * 8;                  // C/D layout: VGPR r, lane half
        size_t idx = (size_t)(tb * CDIM + o) * NDIM + n;
        float sq = qg[o] * rsqrtf(qv[o] + BN_EPS);
        float sk = kg[o] * rsqrtf(kv_[o] + BN_EPS);
        float sv = vg[o] * rsqrtf(vv[o] + BN_EPS);
        qcur[idx] = aq[r] * sq + (qb[o] - qm[o] * sq);
        kcur[idx] = ak[r] * sk + (kb[o] - km[o] * sk);
        vcur[idx] = av[r] * sv + (vb[o] - vm[o] * sv);
    }
}

// --- K3: branch LIFs (thresh 1.0); v-spikes also emitted to d_out ---------
__global__ void k_lif_branches(const float* __restrict__ qcur,
                               const float* __restrict__ kcur,
                               const float* __restrict__ vcur,
                               _Float16* __restrict__ qs, _Float16* __restrict__ ks,
                               _Float16* __restrict__ vs, float* __restrict__ vout) {
    int i = blockIdx.x * blockDim.x + threadIdx.x;     // over (b,c,n)
    int b = i >> 17, c = (i >> 8) & 511, n = i & 255;
    int h = c >> 6, d = c & 63;
    const float* curs[3] = {qcur, kcur, vcur};
    _Float16* sps[3] = {qs, ks, vs};
#pragma unroll
    for (int br = 0; br < 3; ++br) {
        float v = 0.f;
#pragma unroll
        for (int t = 0; t < TDIM; ++t) {
            float y = curs[br][(size_t)t * BCN + i];
            v += (y - v) * 0.5f;
            float s = (v >= 1.0f) ? 1.0f : 0.0f;
            v *= (1.0f - s);
            sps[br][(size_t)t * BCN + i] = (_Float16)s;
            if (br == 2) {                             // v output: (t,b,h,n,d)
                size_t vi = ((size_t)((t * BDIM + b) * HEADS + h) * NDIM + n) * DH + d;
                vout[vi] = s;
            }
        }
    }
}

// --- K4: kv = sum_n k*v ; one wave per (t,b,c) row -------------------------
__global__ void k_kv_reduce(const _Float16* __restrict__ ks,
                            const _Float16* __restrict__ vs,
                            float* __restrict__ kvcur) {
    int lane = threadIdx.x & 31;
    int row  = blockIdx.x * 8 + (threadIdx.x >> 5);    // T*B*C = 32768 rows
    const v8h kh = *(const v8h*)(ks + (size_t)row * NDIM + lane * 8);
    const v8h vh = *(const v8h*)(vs + (size_t)row * NDIM + lane * 8);
    float s = 0.f;
#pragma unroll
    for (int j = 0; j < 8; ++j) s += (float)kh[j] * (float)vh[j];
#pragma unroll
    for (int off = 16; off > 0; off >>= 1) s += __shfl_xor(s, off, 32);
    if (lane == 0) kvcur[row] = s;
}

// --- K5: talking-heads LIF on kv (thresh 0.5) ------------------------------
__global__ void k_lif_kv(const float* __restrict__ kvcur, float* __restrict__ kvs) {
    int i = blockIdx.x * blockDim.x + threadIdx.x;     // B*C = 8192
    float v = 0.f;
#pragma unroll
    for (int t = 0; t < TDIM; ++t) {
        float y = kvcur[t * (BDIM * CDIM) + i];
        v += (y - v) * 0.5f;
        float s = (v >= 0.5f) ? 1.0f : 0.0f;
        v *= (1.0f - s);
        kvs[t * (BDIM * CDIM) + i] = s;
    }
}

// --- K6: out = q * kv, swizzled straight into B-tile order for proj GEMM ---
__global__ void k_pack_out(const _Float16* __restrict__ qs,
                           const float* __restrict__ kvs,
                           _Float16* __restrict__ outp) {
    int i = blockIdx.x * blockDim.x + threadIdx.x;     // (tb,c,n) = 2^23
    int tb = i >> 17, c = (i >> 8) & 511, n = i & 255;
    float prod = (float)qs[i] * kvs[tb * CDIM + c];
    outp[pack_b_idx(tb, c, n)] = (_Float16)prod;
}

// --- K7: projection GEMM (LDS-staged B) + bias + BN + residual -------------
__global__ void __launch_bounds__(256)
k_gemm_proj(const _Float16* __restrict__ outp,
            const _Float16* __restrict__ wpack,
            const float* __restrict__ bp,
            const float* __restrict__ pg, const float* __restrict__ pb,
            const float* __restrict__ pm, const float* __restrict__ pv,
            const float* __restrict__ x, float* __restrict__ out) {
    __shared__ _Float16 Bs[16 * 512];                  // 16KB B-panel
    int lane = threadIdx.x & 31;
    int wave = threadIdx.x >> 5;
    int blk  = blockIdx.x;
    int mgrp = blk & 3, nt = (blk >> 2) & 15, tb = blk >> 6;
    int mt   = mgrp * 8 + wave;

    {
        const v16h* src = (const v16h*)outp + (size_t)((tb * 16 + nt) * 16) * 32;
        v16h* dst = (v16h*)Bs;
        dst[threadIdx.x]       = src[threadIdx.x];
        dst[threadIdx.x + 256] = src[threadIdx.x + 256];
    }
    __syncthreads();

    const v16h* Bl = (const v16h*)Bs + lane;
    const v16h* Ap = (const v16h*)wpack + (size_t)3 * 16384 + (size_t)(mt * 16) * 32 + lane;

    v8f acc = {};
#pragma unroll 4
    for (int kk = 0; kk < 16; ++kk) {
        v16h btile = Bl[kk * 32];
        if (kk + 1 < 16) __builtin_prefetch((const void*)(Ap + (kk + 1) * 32), 0, 1);
        v16h a = Ap[kk * 32];
        acc = __builtin_amdgcn_wmma_f32_16x16x32_f16(false, a, false, btile, (short)0, acc, false, false);
    }
    int col = lane & 15, hi = lane >> 4;
    int n = nt * 16 + col;
#pragma unroll
    for (int r = 0; r < 8; ++r) {
        int o = mt * 16 + r + hi * 8;
        size_t idx = (size_t)(tb * CDIM + o) * NDIM + n;
        float sc = pg[o] * rsqrtf(pv[o] + BN_EPS);
        float y  = acc[r] + bp[o];
        out[idx] = (y - pm[o]) * sc + pb[o] + x[idx];
    }
}

// ---------------------------------------------------------------------------
extern "C" void kernel_launch(void* const* d_in, const int* in_sizes, int n_in,
                              void* d_out, int out_size, void* d_ws, size_t ws_size,
                              hipStream_t stream) {
    (void)in_sizes; (void)n_in; (void)out_size; (void)ws_size;
    const float* x   = (const float*)d_in[0];
    const float* wq  = (const float*)d_in[1];
    const float *qg = (const float*)d_in[2],  *qb = (const float*)d_in[3];
    const float *qm = (const float*)d_in[4],  *qvar = (const float*)d_in[5];
    const float* wk  = (const float*)d_in[6];
    const float *kg = (const float*)d_in[7],  *kb = (const float*)d_in[8];
    const float *km = (const float*)d_in[9],  *kvar = (const float*)d_in[10];
    const float* wv  = (const float*)d_in[11];
    const float *vg = (const float*)d_in[12], *vb = (const float*)d_in[13];
    const float *vm = (const float*)d_in[14], *vvar = (const float*)d_in[15];
    const float* wp  = (const float*)d_in[16];
    const float* bp  = (const float*)d_in[17];
    const float *pg = (const float*)d_in[18], *pb = (const float*)d_in[19];
    const float *pm = (const float*)d_in[20], *pvar = (const float*)d_in[21];

    char* ws = (char*)d_ws;
    _Float16* xs_pack = (_Float16*)(ws + XS_PACK_OFF);
    _Float16* wpack   = (_Float16*)(ws + WPACK_OFF);
    float*    qcur    = (float*)(ws + QCUR_OFF);
    float*    kcur    = (float*)(ws + KCUR_OFF);
    float*    vcur    = (float*)(ws + VCUR_OFF);
    _Float16* qs      = (_Float16*)(ws + QS_OFF);
    _Float16* ks      = (_Float16*)(ws + KS_OFF);
    _Float16* vs      = (_Float16*)(ws + VS_OFF);
    float*    kvcur   = (float*)(ws + KVCUR_OFF);
    float*    kvs     = (float*)(ws + KVS_OFF);
    _Float16* outpack = (_Float16*)(ws + OUTPACK_OFF);   // aliases kcur (dead)

    float* out  = (float*)d_out;                         // (T,B,C,H,W)
    float* vout = out + TBCN;                            // (T,B,h,N,dh)

    k_pack_weights<<<4096, 256, 0, stream>>>(wq, wk, wv, wp, wpack);
    k_lif_shortcut<<<BCN / 256, 256, 0, stream>>>(x, xs_pack);
    k_gemm_qkv<<<4096, 256, 0, stream>>>(xs_pack, wpack,
                                         qg, qb, qm, qvar,
                                         kg, kb, km, kvar,
                                         vg, vb, vm, vvar,
                                         qcur, kcur, vcur);
    k_lif_branches<<<BCN / 256, 256, 0, stream>>>(qcur, kcur, vcur, qs, ks, vs, vout);
    k_kv_reduce<<<4096, 256, 0, stream>>>(ks, vs, kvcur);
    k_lif_kv<<<(BDIM * CDIM) / 256, 256, 0, stream>>>(kvcur, kvs);
    k_pack_out<<<TBCN / 256, 256, 0, stream>>>(qs, kvs, outpack);
    k_gemm_proj<<<4096, 256, 0, stream>>>(outpack, wpack, bp, pg, pb, pm, pvar, x, out);
}
